// KeopsBlockAttentionProduct_72121090834677
// MI455X (gfx1250) — matrix operand order repaired and checked by reading, not compiled, not run on hardware
//
#include <hip/hip_runtime.h>

typedef _Float16 v16h  __attribute__((ext_vector_type(16)));
typedef _Float16 f16x2 __attribute__((ext_vector_type(2)));
typedef float    v8f   __attribute__((ext_vector_type(8)));

#define TT   8192
#define DD   64
#define WIN  256
#define SPAN 512
#define OFF  128
#define NB   (TT / WIN)   // 32
#define KC   128          // keys per LDS chunk
#define NEGINF (-1.0e30f)

// 1/sqrt(D) * log2(e): scores land directly in log2 domain, so softmax uses raw v_exp_f32
#define QSCALE (0.125f * 1.44269504088896340736f)

#define KLDS_STRIDE 66    // 64 + 2 pad halves  (odd dword stride -> conflict-free)
#define VLDS_STRIDE 130   // 128 + 2 pad halves

__global__ __launch_bounds__(256)
void swin_flash_attn_wmma(const float* __restrict__ Q,
                          const float* __restrict__ K,
                          const float* __restrict__ V,
                          float* __restrict__ Out)
{
  __shared__ __align__(16) _Float16 Klds[KC * KLDS_STRIDE]; // K chunk, [key][d]
  __shared__ __align__(16) _Float16 Vtl [DD * VLDS_STRIDE]; // V chunk transposed, [d][key]

  const int g    = blockIdx.x;
  const int nblk = g & (NB - 1);
  const int bh   = g >> 5;          // NB == 32
  const int tid  = threadIdx.x;
  const int lane = tid & 31;
  const int wv   = tid >> 5;        // wave id 0..7  (wave32)
  const int l16  = lane & 15;
  const int h16  = lane >> 4;

  const size_t bhBase = (size_t)bh * TT * DD;
  const float* Qg = Q + bhBase + (size_t)nblk * WIN * DD;
  const float* Kg = K + bhBase;
  const float* Vg = V + bhBase;
  float*       Og = Out + bhBase + (size_t)nblk * WIN * DD;

  // ---- Q fragments as WMMA B operands (k = d dim, n = query), pre-scaled ----
  // B 32x16 f16, contiguous k order used here (matches A): lane n = l16, K = 16*h16 + {2v,2v+1}
  v16h bq[2][2];
  #pragma unroll
  for (int nt = 0; nt < 2; ++nt) {
    const float* qrow = Qg + (size_t)(wv * 32 + nt * 16 + l16) * DD;
    #pragma unroll
    for (int kf = 0; kf < 2; ++kf) {
      v16h bb;
      #pragma unroll
      for (int v = 0; v < 8; ++v) {
        const int d = kf * 32 + 16 * h16 + 2 * v;
        const float2 f = *(const float2*)(qrow + d);
        bb[2 * v]     = (_Float16)(f.x * QSCALE);
        bb[2 * v + 1] = (_Float16)(f.y * QSCALE);
      }
      bq[nt][kf] = bb;
    }
  }

  const v8f vzero = {0.f, 0.f, 0.f, 0.f, 0.f, 0.f, 0.f, 0.f};
  v8f acc[2][4];                 // O accumulators: [qtile][dtile], rows=q, cols=d
  #pragma unroll
  for (int nt = 0; nt < 2; ++nt)
    #pragma unroll
    for (int ntd = 0; ntd < 4; ++ntd) acc[nt][ntd] = vzero;

  float mrun[2] = { -3.0e4f, -3.0e4f };   // finite sentinel: exp2 underflows to exact 0
  float lrun[2] = { 0.f, 0.f };

  const int j00 = nblk * WIN - OFF;

  for (int c = 0; c < SPAN / KC; ++c) {
    if (c) __syncthreads();
    const int j0 = j00 + c * KC;
    // workgroup-uniform: does this chunk touch out-of-range keys? (128 of 8192 instances)
    const bool edge = (j0 < 0) || (j0 + KC > TT);

    { // ---- cooperative stage of K (row-major f16) and V (transposed f16) ----
      const int r  = tid >> 1;            // chunk row 0..127
      const int cs = (tid & 1) * 32;      // d segment
      const int j  = j0 + r;
      const int jc = j < 0 ? 0 : (j > TT - 1 ? TT - 1 : j);   // clamp like reference
      const float4* kp = (const float4*)(Kg + (size_t)jc * DD + cs);
      const float4* vp = (const float4*)(Vg + (size_t)jc * DD + cs);
      #pragma unroll
      for (int i = 0; i < 8; ++i) {
        const float4 kx = kp[i];
        const float4 vx = vp[i];
        const int d = cs + i * 4;
        _Float16* kd = &Klds[r * KLDS_STRIDE + d];
        f16x2 p01; p01[0] = (_Float16)kx.x; p01[1] = (_Float16)kx.y;
        f16x2 p23; p23[0] = (_Float16)kx.z; p23[1] = (_Float16)kx.w;
        *(f16x2*)(kd)     = p01;
        *(f16x2*)(kd + 2) = p23;
        Vtl[(d + 0) * VLDS_STRIDE + r] = (_Float16)vx.x;
        Vtl[(d + 1) * VLDS_STRIDE + r] = (_Float16)vx.y;
        Vtl[(d + 2) * VLDS_STRIDE + r] = (_Float16)vx.z;
        Vtl[(d + 3) * VLDS_STRIDE + r] = (_Float16)vx.w;
      }
      if (c + 1 < SPAN / KC) {            // prefetch next chunk -> global_prefetch_b8
        const int jn  = j + KC;
        const int jnc = jn < 0 ? 0 : (jn > TT - 1 ? TT - 1 : jn);
        __builtin_prefetch(Kg + (size_t)jnc * DD + cs, 0, 1);
        __builtin_prefetch(Vg + (size_t)jnc * DD + cs, 0, 1);
      }
    }
    __syncthreads();

    #pragma unroll
    for (int grp = 0; grp < KC / 32; ++grp) {
      const int kb = grp * 32;

      // ---- V B-fragments (k = key, n = d), pairs contiguous in Vtl ----
      v16h bv[4];
      #pragma unroll
      for (int ntd = 0; ntd < 4; ++ntd) {
        const _Float16* vr = &Vtl[(ntd * 16 + l16) * VLDS_STRIDE + kb + 8 * h16];
        v16h bb;
        #pragma unroll
        for (int v = 0; v < 8; ++v) {
          const f16x2 p = *(const f16x2*)(vr + ((v < 4) ? 2 * v : 2 * v + 8));
          bb[2 * v] = p[0]; bb[2 * v + 1] = p[1];
        }
        bv[ntd] = bb;
      }

      // ---- S^T = K * Q^T : 2 key-tiles x 2 query-tiles ----
      v8f st[2][2];
      #pragma unroll
      for (int t = 0; t < 2; ++t) {
        const _Float16* kr = &Klds[(kb + t * 16 + l16) * KLDS_STRIDE];
        v16h a0, a1;     // A 16x32: lane m = key, K = 16*h16 + s (contiguous)
        #pragma unroll
        for (int v = 0; v < 8; ++v) {
          const int d = 2 * v + 16 * h16;
          const f16x2 p0 = *(const f16x2*)(kr + d);
          const f16x2 p1 = *(const f16x2*)(kr + d + 32);
          a0[2 * v] = p0[0]; a0[2 * v + 1] = p0[1];
          a1[2 * v] = p1[0]; a1[2 * v + 1] = p1[1];
        }
        #pragma unroll
        for (int nt = 0; nt < 2; ++nt) {
          v8f cf = __builtin_amdgcn_wmma_f32_16x16x32_f16(false, a0, false, bq[nt][0],
                                                          (short)0, vzero, false, false);
          cf     = __builtin_amdgcn_wmma_f32_16x16x32_f16(false, a1, false, bq[nt][1],
                                                          (short)0, cf, false, false);
          st[t][nt] = cf;
        }
      }

      // ---- mask out-of-range keys: only ever taken on sequence-edge chunks ----
      if (edge) {
        const int jbase = j0 + kb + 8 * h16;
        #pragma unroll
        for (int t = 0; t < 2; ++t)
          #pragma unroll
          for (int v = 0; v < 8; ++v) {
            const int j = jbase + t * 16 + v;
            if (j < 0 || j >= TT) { st[t][0][v] = NEGINF; st[t][1][v] = NEGINF; }
          }
      }

      // ---- online softmax in log2 domain (per-lane along keys) + P*V ----
      #pragma unroll
      for (int nt = 0; nt < 2; ++nt) {
        float tmax = NEGINF;
        #pragma unroll
        for (int t = 0; t < 2; ++t)
          #pragma unroll
          for (int v = 0; v < 8; ++v) tmax = fmaxf(tmax, st[t][nt][v]);
        tmax = fmaxf(tmax, __shfl_xor(tmax, 16, 32));   // merge lane-halves (shared max)

        const float mnew = fmaxf(mrun[nt], tmax);
        const float sc   = __builtin_amdgcn_exp2f(mrun[nt] - mnew);   // raw v_exp_f32
        mrun[nt] = mnew;

        float psum = 0.f;
        #pragma unroll
        for (int t = 0; t < 2; ++t)
          #pragma unroll
          for (int v = 0; v < 8; ++v) {
            const float p = __builtin_amdgcn_exp2f(st[t][nt][v] - mnew);
            st[t][nt][v] = p;
            psum += p;
          }
        lrun[nt] = lrun[nt] * sc + psum;

        // rescale O rows (row q lives in lane q of the softmax state)
        #pragma unroll
        for (int v = 0; v < 8; ++v) {
          const float rs = __shfl(sc, v + 8 * h16, 32);
          #pragma unroll
          for (int ntd = 0; ntd < 4; ++ntd) acc[nt][ntd][v] *= rs;
        }

        // repack P (C-layout of S^T) into WMMA A operand, fully in-register
        v16h ap;
        #pragma unroll
        for (int v = 0; v < 4; ++v) {
          ap[2 * v]     = (_Float16)st[0][nt][2 * v];
          ap[2 * v + 1] = (_Float16)st[0][nt][2 * v + 1];
          ap[2 * v + 8] = (_Float16)st[1][nt][2 * v];
          ap[2 * v + 9] = (_Float16)st[1][nt][2 * v + 1];
        }
        #pragma unroll
        for (int ntd = 0; ntd < 4; ++ntd)
          acc[nt][ntd] = __builtin_amdgcn_wmma_f32_16x16x32_f16(false, ap, false, bv[ntd],
                                                                (short)0, acc[nt][ntd],
                                                                false, false);
      }
    }
  }

  // ---- normalize by row sums and store fp32 ----
  #pragma unroll
  for (int nt = 0; nt < 2; ++nt) {
    const float lt   = lrun[nt] + __shfl_xor(lrun[nt], 16, 32);
    const float linv = 1.0f / lt;
    #pragma unroll
    for (int v = 0; v < 8; ++v) {
      const float rs = __shfl(linv, v + 8 * h16, 32);
      float* orow = Og + (size_t)(wv * 32 + nt * 16 + v + 8 * h16) * DD;
      #pragma unroll
      for (int ntd = 0; ntd < 4; ++ntd)
        orow[ntd * 16 + l16] = acc[nt][ntd][v] * rs;
    }
  }
}

extern "C" void kernel_launch(void* const* d_in, const int* in_sizes, int n_in,
                              void* d_out, int out_size, void* d_ws, size_t ws_size,
                              hipStream_t stream) {
  (void)in_sizes; (void)n_in; (void)out_size; (void)d_ws; (void)ws_size;
  const float* Q = (const float*)d_in[0];
  const float* K = (const float*)d_in[1];
  const float* V = (const float*)d_in[2];
  float* O = (float*)d_out;
  const int B = 4, H = 16;
  dim3 grid(B * H * NB);   // 2048 workgroups, one per (b,h,query-block)
  dim3 block(256);         // 8 wave32 waves
  hipLaunchKernelGGL(swin_flash_attn_wmma, grid, block, 0, stream, Q, K, V, O);
}
